// MaxSimilarity_52140902973492
// MI455X (gfx1250) — matrix operand order repaired and checked by reading, not compiled, not run on hardware
//
#include <hip/hip_runtime.h>
#include <hip/hip_bf16.h>

// ---------------------------------------------------------------------------
// MaxSimilarity on MI455X (gfx1250, wave32):
//   out[i] = max_j  (x1[i,:] . x2[j,:]) / (||x1[i]|| * ||x2[j]||)
// Row-normalize, split to bf16 hi/lo, GEMM via v_wmma_f32_16x16x32_bf16
// (hi*hi + hi*lo + lo*hi ~ f32 accuracy), fused running row-max.
// Data movement: global_load_async_to_lds_b128 with double-buffered LDS.
// ---------------------------------------------------------------------------

typedef __attribute__((ext_vector_type(16))) __bf16 v16bf;
typedef __attribute__((ext_vector_type(8)))  float  v8f;

#define N1 8192
#define N2 16384
#define KDIM 1024

#define BM 128          // rows per workgroup
#define BN 128          // cols per n-tile iteration
#define NBLK_COLS 2048  // cols per workgroup (per blockIdx.y)
#define BK 32           // K per WMMA step
#define KTILES (KDIM / BK)
#define LDS_STRIDE 80   // padded row stride (32 bf16 = 64B + 16B pad): bank-conflict-free b128
#define TILE_BYTES (BM * LDS_STRIDE)   // 10240 B per tile
#define STAGE_BYTES (4 * TILE_BYTES)   // A-hi, A-lo, B-hi, B-lo

// ws layout (bytes)
#define OFF_X1HI 0
#define OFF_X1LO (16777216)
#define OFF_X2HI (33554432)
#define OFF_X2LO (67108864)
#define OFF_PART (100663296)

__device__ __forceinline__ unsigned short f2bf_rn(float f) {
  unsigned int u = __float_as_uint(f);
  unsigned int r = u + 0x7FFFu + ((u >> 16) & 1u);  // round to nearest even
  return (unsigned short)(r >> 16);
}
__device__ __forceinline__ float bf2f(unsigned short h) {
  return __uint_as_float(((unsigned int)h) << 16);
}

// Flat shared pointer -> 32-bit LDS byte address (low 32 bits of the pointer).
__device__ __forceinline__ unsigned lds_lo32(const void* p) {
  return (unsigned)(unsigned long long)p;
}

// CDNA5 async copy: global -> LDS at the vdst address, 16B per lane, tracked by ASYNCcnt.
__device__ __forceinline__ void async_ld_b128(unsigned lds_byte, const void* g) {
  asm volatile("global_load_async_to_lds_b128 %0, %1, off"
               :: "v"(lds_byte), "v"(g)
               : "memory");
}
__device__ __forceinline__ void wait_async0() {
  asm volatile("s_wait_asynccnt 0x0" ::: "memory");
}

// ---------------------------------------------------------------------------
// Kernel 1: per-row L2 norm, normalize, split into bf16 hi + bf16 lo.
// One 256-thread block per row (K=1024 -> 4 floats/thread).
// ---------------------------------------------------------------------------
__global__ __launch_bounds__(256) void MaxSim_prep_kernel(
    const float* __restrict__ x1, const float* __restrict__ x2,
    unsigned short* __restrict__ x1hi, unsigned short* __restrict__ x1lo,
    unsigned short* __restrict__ x2hi, unsigned short* __restrict__ x2lo) {
  const int row = blockIdx.x;
  const int t = threadIdx.x;

  const float* src;
  unsigned short *dh, *dl;
  if (row < N1) {
    src = x1 + (size_t)row * KDIM;
    dh = x1hi + (size_t)row * KDIM;
    dl = x1lo + (size_t)row * KDIM;
  } else {
    const int r2 = row - N1;
    src = x2 + (size_t)r2 * KDIM;
    dh = x2hi + (size_t)r2 * KDIM;
    dl = x2lo + (size_t)r2 * KDIM;
  }

  float4 v = *(const float4*)(src + t * 4);
  float s = v.x * v.x + v.y * v.y + v.z * v.z + v.w * v.w;
  // wave32 reduction
  s += __shfl_xor(s, 16, 32);
  s += __shfl_xor(s, 8, 32);
  s += __shfl_xor(s, 4, 32);
  s += __shfl_xor(s, 2, 32);
  s += __shfl_xor(s, 1, 32);
  __shared__ float red[8];
  if ((t & 31) == 0) red[t >> 5] = s;
  __syncthreads();
  float tot = red[0] + red[1] + red[2] + red[3] +
              red[4] + red[5] + red[6] + red[7];
  float rn = (tot > 1e-24f) ? rsqrtf(tot) : 0.0f;

  float y[4] = {v.x * rn, v.y * rn, v.z * rn, v.w * rn};
  ushort4 hv, lv;
  unsigned short* hp = &hv.x;
  unsigned short* lp = &lv.x;
#pragma unroll
  for (int i = 0; i < 4; ++i) {
    unsigned short h = f2bf_rn(y[i]);
    hp[i] = h;
    lp[i] = f2bf_rn(y[i] - bf2f(h));
  }
  *(ushort4*)(dh + t * 4) = hv;
  *(ushort4*)(dl + t * 4) = lv;
}

// ---------------------------------------------------------------------------
// Kernel 2: fused bf16x3 GEMM + running row-max.
// Grid: (64 M-blocks, 8 N-blocks). 256 threads = 8 wave32s.
// Each wave computes a 16(M) x 128(N) strip: 8 WMMA accumulators.
// K-slices are staged with async-to-LDS copies into a double buffer.
// ---------------------------------------------------------------------------
__device__ __forceinline__ void stage_async(
    unsigned char* sbase,
    const unsigned char* x1hi, const unsigned char* x1lo,
    const unsigned char* x2hi, const unsigned char* x2lo,
    int t, int mBase, int nBase, int kByte) {
#pragma unroll
  for (int i = 0; i < 2; ++i) {
    const int c = t + i * 256;       // 512 x 16B chunks per tile
    const int row = c >> 2;
    const int c4 = c & 3;
    const int lofs = row * LDS_STRIDE + c4 * 16;
    const size_t ga = (size_t)(mBase + row) * (KDIM * 2) + kByte + c4 * 16;
    const size_t gb = (size_t)(nBase + row) * (KDIM * 2) + kByte + c4 * 16;
    async_ld_b128(lds_lo32(sbase + lofs),                  x1hi + ga);
    async_ld_b128(lds_lo32(sbase + TILE_BYTES + lofs),     x1lo + ga);
    async_ld_b128(lds_lo32(sbase + 2 * TILE_BYTES + lofs), x2hi + gb);
    async_ld_b128(lds_lo32(sbase + 3 * TILE_BYTES + lofs), x2lo + gb);
  }
}

__global__ __launch_bounds__(256) void MaxSim_gemm_kernel(
    const unsigned char* __restrict__ x1hi, const unsigned char* __restrict__ x1lo,
    const unsigned char* __restrict__ x2hi, const unsigned char* __restrict__ x2lo,
    float* __restrict__ partial) {
  __shared__ __align__(16) unsigned char smem[2 * STAGE_BYTES];  // 80 KB

  const int t = threadIdx.x;
  const int lane = t & 31;
  const int w = t >> 5;        // wave id 0..7
  const int h = lane >> 4;     // half-wave
  const int lr = lane & 15;    // lane within half

  const int mBase = blockIdx.x * BM;
  const int nBase0 = blockIdx.y * NBLK_COLS;

  float rm[8];
#pragma unroll
  for (int r = 0; r < 8; ++r) rm[r] = -__builtin_inff();

  for (int nt = 0; nt < NBLK_COLS / BN; ++nt) {
    const int nBase = nBase0 + nt * BN;
    v8f acc[8] = {};

    stage_async(smem, x1hi, x1lo, x2hi, x2lo, t, mBase, nBase, 0);

    for (int kt = 0; kt < KTILES; ++kt) {
      unsigned char* cur = smem + (kt & 1) * STAGE_BYTES;
      unsigned char* nxt = smem + ((kt + 1) & 1) * STAGE_BYTES;

      wait_async0();      // this wave's copies into `cur` have landed
      __syncthreads();    // everyone's copies visible to everyone

      if (kt + 1 < KTILES)  // overlap next K-slice copy with this slice's math
        stage_async(nxt, x1hi, x1lo, x2hi, x2lo, t, mBase, nBase,
                    (kt + 1) * (BK * 2));

      // A fragment (16x32 bf16): lane lr = row, halves pick K {0..15}
      // (h=0) or {16..31} (h=1): chunks at bytes 32h and 32h+16.
      v16bf ahi, alo;
      {
        const int ro = (16 * w + lr) * LDS_STRIDE;
        uint4* p = (uint4*)&ahi;
        p[0] = *(const uint4*)(cur + ro + 32 * h);
        p[1] = *(const uint4*)(cur + ro + 32 * h + 16);
        uint4* q = (uint4*)&alo;
        q[0] = *(const uint4*)(cur + TILE_BYTES + ro + 32 * h);
        q[1] = *(const uint4*)(cur + TILE_BYTES + ro + 32 * h + 16);
      }

      // Process accumulators in pairs so back-to-back WMMAs are independent.
#pragma unroll
      for (int ct = 0; ct < 8; ct += 2) {
        const int ro0 = (16 * ct + lr) * LDS_STRIDE + 32 * h;
        const int ro1 = (16 * (ct + 1) + lr) * LDS_STRIDE + 32 * h;
        v16bf bhi0, blo0, bhi1, blo1;
        uint4* p;
        p = (uint4*)&bhi0;
        p[0] = *(const uint4*)(cur + 2 * TILE_BYTES + ro0);
        p[1] = *(const uint4*)(cur + 2 * TILE_BYTES + ro0 + 16);
        p = (uint4*)&blo0;
        p[0] = *(const uint4*)(cur + 3 * TILE_BYTES + ro0);
        p[1] = *(const uint4*)(cur + 3 * TILE_BYTES + ro0 + 16);
        p = (uint4*)&bhi1;
        p[0] = *(const uint4*)(cur + 2 * TILE_BYTES + ro1);
        p[1] = *(const uint4*)(cur + 2 * TILE_BYTES + ro1 + 16);
        p = (uint4*)&blo1;
        p[0] = *(const uint4*)(cur + 3 * TILE_BYTES + ro1);
        p[1] = *(const uint4*)(cur + 3 * TILE_BYTES + ro1 + 16);

        acc[ct]     = __builtin_amdgcn_wmma_f32_16x16x32_bf16(
            false, ahi, false, bhi0, (short)0, acc[ct],     false, false);
        acc[ct + 1] = __builtin_amdgcn_wmma_f32_16x16x32_bf16(
            false, ahi, false, bhi1, (short)0, acc[ct + 1], false, false);
        acc[ct]     = __builtin_amdgcn_wmma_f32_16x16x32_bf16(
            false, ahi, false, blo0, (short)0, acc[ct],     false, false);
        acc[ct + 1] = __builtin_amdgcn_wmma_f32_16x16x32_bf16(
            false, ahi, false, blo1, (short)0, acc[ct + 1], false, false);
        acc[ct]     = __builtin_amdgcn_wmma_f32_16x16x32_bf16(
            false, alo, false, bhi0, (short)0, acc[ct],     false, false);
        acc[ct + 1] = __builtin_amdgcn_wmma_f32_16x16x32_bf16(
            false, alo, false, bhi1, (short)0, acc[ct + 1], false, false);
      }
      __syncthreads();    // all reads of `cur` done before it is refilled
    }

    // Fold this n-tile's sims into the running per-row max.
    // C layout: reg r holds (row = r + 8h, col = lr) of the 16x16 tile.
#pragma unroll
    for (int ct = 0; ct < 8; ++ct)
#pragma unroll
      for (int r = 0; r < 8; ++r) rm[r] = fmaxf(rm[r], acc[ct][r]);
  }

  // Reduce across the 16 lanes that share a row (xor 1,2,4,8 stays in-half).
#pragma unroll
  for (int r = 0; r < 8; ++r) {
    float v = rm[r];
    v = fmaxf(v, __shfl_xor(v, 1, 32));
    v = fmaxf(v, __shfl_xor(v, 2, 32));
    v = fmaxf(v, __shfl_xor(v, 4, 32));
    v = fmaxf(v, __shfl_xor(v, 8, 32));
    rm[r] = v;
  }
  if (lr == 0) {
#pragma unroll
    for (int r = 0; r < 8; ++r) {
      const int row = mBase + 16 * w + r + 8 * h;
      partial[(size_t)blockIdx.y * N1 + row] = rm[r];
    }
  }
}

// ---------------------------------------------------------------------------
// Kernel 3: reduce 8 N-block partial maxima per row.
// ---------------------------------------------------------------------------
__global__ __launch_bounds__(256) void MaxSim_final_kernel(
    const float* __restrict__ partial, float* __restrict__ out) {
  const int i = blockIdx.x * 256 + threadIdx.x;
  if (i < N1) {
    float m = partial[i];
#pragma unroll
    for (int y = 1; y < N2 / NBLK_COLS; ++y)
      m = fmaxf(m, partial[(size_t)y * N1 + i]);
    out[i] = m;
  }
}

extern "C" void kernel_launch(void* const* d_in, const int* in_sizes, int n_in,
                              void* d_out, int out_size, void* d_ws, size_t ws_size,
                              hipStream_t stream) {
  const float* x1 = (const float*)d_in[0];
  const float* x2 = (const float*)d_in[1];
  float* out = (float*)d_out;
  unsigned char* ws = (unsigned char*)d_ws;

  unsigned short* x1hi = (unsigned short*)(ws + OFF_X1HI);
  unsigned short* x1lo = (unsigned short*)(ws + OFF_X1LO);
  unsigned short* x2hi = (unsigned short*)(ws + OFF_X2HI);
  unsigned short* x2lo = (unsigned short*)(ws + OFF_X2LO);
  float* partial = (float*)(ws + OFF_PART);

  MaxSim_prep_kernel<<<N1 + N2, 256, 0, stream>>>(x1, x2, x1hi, x1lo, x2hi, x2lo);

  dim3 grid(N1 / BM, N2 / NBLK_COLS);  // (64, 8)
  MaxSim_gemm_kernel<<<grid, 256, 0, stream>>>(
      (const unsigned char*)x1hi, (const unsigned char*)x1lo,
      (const unsigned char*)x2hi, (const unsigned char*)x2lo, partial);

  MaxSim_final_kernel<<<N1 / 256, 256, 0, stream>>>(partial, out);
}
